// HistoryEncoder_81552839016996
// MI455X (gfx1250) — matrix operand order repaired and checked by reading, not compiled, not run on hardware
//
#include <hip/hip_runtime.h>
#include <hip/hip_bf16.h>

typedef __attribute__((ext_vector_type(16))) _Float16 v16h;
typedef __attribute__((ext_vector_type(8)))  float    v8f;

#define NROWS   1000000
#define MLOC    100000
#define NB      16900000            // 169 * 100000 hash buckets
#define OUTD    128

#define SCAN_TILE   1024
#define SCAN_NBLK   ((NB + SCAN_TILE - 1) / SCAN_TILE)      // 16504
#define SCAN2_NCHUNK ((SCAN_NBLK + 255) / 256)              // 65

// d_ws byte offsets (total ~75.7 MB)
#define OFF_NU    0
#define OFF_GID   256
#define OFF_GCNT  (OFF_GID  + (size_t)NB * 4)               // 67,600,256
#define OFF_BSUM  (OFF_GCNT + (size_t)NROWS * 4)            // 71,600,256
#define OFF_GRPH  (OFF_BSUM + (size_t)((SCAN_NBLK * 4 + 255) / 256) * 256)
#define OFF_WT    (OFF_GRPH + (size_t)NROWS * 4)

// ---------------- zero scratch (presence buckets + per-group counters) ----------
__global__ void k_zero(uint4* p, unsigned n4) {
  unsigned i = blockIdx.x * blockDim.x + threadIdx.x;
  if (i < n4) { uint4 z = {0u, 0u, 0u, 0u}; p[i] = z; }
}

// ---------------- zero the u_sum accumulation columns of d_out ------------------
__global__ void k_zero_acc(float* out) {
  unsigned i = blockIdx.x * blockDim.x + threadIdx.x;   // 8M threads, exact
  unsigned r = i >> 3, j = i & 7;
  float4 z = {0.f, 0.f, 0.f, 0.f};
  *(float4*)(out + (size_t)r * OUTD + 96 + j * 4) = z;
}

// ---------------- W (f32, row-major [o][k]) -> f16 ------------------------------
__global__ void k_wt(const float* __restrict__ W, _Float16* __restrict__ Wt) {
  unsigned i = blockIdx.x * 256 + threadIdx.x;
  Wt[i] = (_Float16)W[i];
}

// ---------------- presence marking (idempotent plain stores) --------------------
__global__ void k_presence(const int* __restrict__ x, const int* __restrict__ t,
                           unsigned* __restrict__ gid) {
  unsigned i = blockIdx.x * blockDim.x + threadIdx.x;
  if (i >= NROWS) return;
  unsigned h = (unsigned)t[i] * MLOC + (unsigned)x[i];
  gid[h] = 1u;
}

// ---------------- scan pass 1: in-place exclusive presence scan per 1024-tile ---
__global__ void k_scan1(unsigned* __restrict__ gid, unsigned* __restrict__ bsum) {
  __shared__ unsigned sd[256];
  unsigned tid  = threadIdx.x;
  unsigned base = blockIdx.x * SCAN_TILE + tid * 4;
  unsigned p[4]; unsigned s = 0;
#pragma unroll
  for (int i = 0; i < 4; ++i) {
    unsigned idx = base + i;
    unsigned v = (idx < NB) ? gid[idx] : 0u;
    p[i] = (v != 0u) ? 1u : 0u;
    s += p[i];
  }
  sd[tid] = s;
  __syncthreads();
  for (int off = 1; off < 256; off <<= 1) {
    unsigned tv = (tid >= (unsigned)off) ? sd[tid - off] : 0u;
    __syncthreads();
    sd[tid] += tv;
    __syncthreads();
  }
  unsigned run = sd[tid] - s;            // exclusive prefix for this thread
#pragma unroll
  for (int i = 0; i < 4; ++i) {
    unsigned idx = base + i;
    if (idx < NB) gid[idx] = run;        // in-place: bucket -> partial rank
    run += p[i];
  }
  if (tid == 0) bsum[blockIdx.x] = sd[255];
}

// ---------------- scan pass 2: scan of block sums + n_unique --------------------
__global__ void k_scan2(unsigned* __restrict__ bsum, unsigned* __restrict__ nu) {
  __shared__ unsigned sd[256];
  __shared__ unsigned carry;
  unsigned tid = threadIdx.x;
  if (tid == 0) carry = 0u;
  __syncthreads();
  for (int c = 0; c < SCAN2_NCHUNK; ++c) {
    unsigned j = (unsigned)c * 256 + tid;
    unsigned s = (j < SCAN_NBLK) ? bsum[j] : 0u;
    sd[tid] = s;
    __syncthreads();
    for (int off = 1; off < 256; off <<= 1) {
      unsigned tv = (tid >= (unsigned)off) ? sd[tid - off] : 0u;
      __syncthreads();
      sd[tid] += tv;
      __syncthreads();
    }
    unsigned tot   = sd[255];
    unsigned basec = carry;
    unsigned excl  = sd[tid] - s;
    if (j < SCAN_NBLK) bsum[j] = basec + excl;
    __syncthreads();
    if (tid == 0) carry = basec + tot;
    __syncthreads();
  }
  if (tid == 0) nu[0] = carry;
}

// ---------------- scatter: user-emb sums into d_out[:,96:128], group metadata ---
__global__ __launch_bounds__(256) void k_scatter(
    const int* __restrict__ x, const int* __restrict__ t, const int* __restrict__ u,
    const float* __restrict__ user_emb, const unsigned* __restrict__ gid,
    const unsigned* __restrict__ bsum, unsigned* __restrict__ grp_h,
    unsigned* __restrict__ grp_cnt, float* __restrict__ out) {
  unsigned tid  = threadIdx.x;
  unsigned r    = blockIdx.x * 8 + (tid >> 5);  // 8 rows per block, exact grid
  unsigned lane = tid & 31;
  unsigned h = (unsigned)t[r] * MLOC + (unsigned)x[r];
  unsigned g = gid[h] + bsum[h >> 10];
  if (lane == 0) { grp_h[g] = h; atomicAdd(&grp_cnt[g], 1u); }
  float ue = user_emb[(unsigned)u[r] * 32 + lane];
  atomicAdd(out + (size_t)g * OUTD + 96 + lane, ue);
}

// Branchless tanh: 1 - 2/(exp(2x)+1). Native v_exp_f32 + v_rcp_f32 (~1 ulp each).
// x=+inf -> exp=inf -> rcp=0 -> +1; x=-inf -> exp=0 -> rcp(1) -> -1. Abs err ~1e-7.
__device__ __forceinline__ float fast_tanh(float xv) {
  float e = __expf(2.0f * xv);
  return 1.0f - 2.0f * __builtin_amdgcn_rcpf(e + 1.0f);
}

// ---------------- GEMM + bias + tanh via v_wmma_f32_16x16x32_f16 ----------------
__global__ __launch_bounds__(256) void k_gemm(
    const float* __restrict__ loc_emb, const float* __restrict__ time_emb,
    const _Float16* __restrict__ Wt, const float* __restrict__ bias,
    const unsigned* __restrict__ grp_h, const unsigned* __restrict__ grp_cnt,
    const unsigned* __restrict__ nu_p, float* __restrict__ out) {
  __shared__ _Float16 Alds[16][136];            // 16 rows x 128 K, padded
  const unsigned nU  = nu_p[0];
  const unsigned g0  = blockIdx.x * 16;
  const unsigned tid = threadIdx.x;

  // Build A tile (xtu in f16): cols 0-63 loc, 64-95 time, 96-127 u_mean
#pragma unroll
  for (int it = 0; it < 8; ++it) {
    unsigned e = (unsigned)it * 256 + tid;
    unsigned m = e >> 7, k = e & 127;
    unsigned g = g0 + m;
    float val = 0.f;
    if (g < nU) {
      unsigned h = grp_h[g];
      if (k < 64) {
        val = loc_emb[(h % MLOC) * 64 + k];
      } else if (k < 96) {
        val = time_emb[(h / MLOC) * 32 + (k - 64)];
      } else {
        float cnt = (float)grp_cnt[g];
        val = out[(size_t)g * OUTD + k] / cnt;   // u_sum staged in d_out
      }
    }
    Alds[m][k] = (_Float16)val;
  }
  __syncthreads();

  const unsigned lane = tid & 31;
  const unsigned w    = tid >> 5;       // wave -> 16-column slice
  const unsigned nn   = lane & 15;      // A row / B column
  const unsigned hi   = lane >> 4;
  const unsigned o    = w * 16 + nn;    // output column
  const float    bb   = bias[o];

  v8f acc = {};
#pragma unroll
  for (int c = 0; c < 4; ++c) {
    const int K0 = c * 32;
    v16h a, bm;
#pragma unroll
    for (int j = 0; j < 8; ++j) {
      // A layout (ISA 16-bit A 16x32 table)
      int ka = K0 + ((j & 3) << 1) + ((j >> 2) << 4) + (int)(hi << 3);
      a[2 * j]     = Alds[nn][ka];
      a[2 * j + 1] = Alds[nn][ka + 1];
      // B layout: same K order as A (ISA 16-bit 16x32 table)
      int kb = K0 + ((j & 3) << 1) + ((j >> 2) << 4) + (int)(hi << 3);
      bm[2 * j]     = Wt[(size_t)o * 128 + kb];      // B[k][n] = W[o][k]
      bm[2 * j + 1] = Wt[(size_t)o * 128 + kb + 1];
    }
    acc = __builtin_amdgcn_wmma_f32_16x16x32_f16(false, a, false, bm,
                                                 (short)0, acc, false, false);
  }

  // C/D layout: lane -> col n, VGPR v -> row v + 8*hi.
  // Branchless epilogue: one base address, immediate offsets, v_cndmask masking.
  const unsigned gbase = g0 + 8u * hi;
  float* po = out + (size_t)gbase * OUTD + o;
#pragma unroll
  for (int v = 0; v < 8; ++v) {
    float val = fast_tanh(acc[v] + bb);
    po[(size_t)v * OUTD] = (gbase + (unsigned)v < nU) ? val : 0.f;
  }
}

extern "C" void kernel_launch(void* const* d_in, const int* in_sizes, int n_in,
                              void* d_out, int out_size, void* d_ws, size_t ws_size,
                              hipStream_t stream) {
  (void)in_sizes; (void)n_in; (void)out_size; (void)ws_size;
  const float* loc_emb  = (const float*)d_in[0];
  const float* time_emb = (const float*)d_in[1];
  const float* user_emb = (const float*)d_in[2];
  const float* W        = (const float*)d_in[3];
  const float* bias     = (const float*)d_in[4];
  const int*   x        = (const int*)d_in[5];
  const int*   t        = (const int*)d_in[6];
  const int*   u        = (const int*)d_in[7];
  float* out = (float*)d_out;
  char*  ws  = (char*)d_ws;

  unsigned* nu   = (unsigned*)(ws + OFF_NU);
  unsigned* gid  = (unsigned*)(ws + OFF_GID);
  unsigned* gcnt = (unsigned*)(ws + OFF_GCNT);
  unsigned* bsum = (unsigned*)(ws + OFF_BSUM);
  unsigned* grph = (unsigned*)(ws + OFF_GRPH);
  _Float16* Wt   = (_Float16*)(ws + OFF_WT);

  const unsigned n4 = (NB + NROWS) / 4;                    // gid+gcnt contiguous
  k_zero    <<<(n4 + 255) / 256, 256, 0, stream>>>((uint4*)gid, n4);
  k_zero_acc<<<(NROWS * 8) / 256, 256, 0, stream>>>(out);
  k_wt      <<<(128 * 128) / 256, 256, 0, stream>>>(W, Wt);
  k_presence<<<(NROWS + 255) / 256, 256, 0, stream>>>(x, t, gid);
  k_scan1   <<<SCAN_NBLK, 256, 0, stream>>>(gid, bsum);
  k_scan2   <<<1, 256, 0, stream>>>(bsum, nu);
  k_scatter <<<NROWS / 8, 256, 0, stream>>>(x, t, u, user_emb, gid, bsum, grph, gcnt, out);
  k_gemm    <<<NROWS / 16, 256, 0, stream>>>(loc_emb, time_emb, Wt, bias, grph, gcnt, nu, out);
}
